// CausalSelfAttention_62234076119542
// MI455X (gfx1250) — matrix operand order repaired and checked by reading, not compiled, not run on hardware
//
#include <hip/hip_runtime.h>
#include <hip/hip_bf16.h>

typedef __bf16 bf16;
typedef __attribute__((ext_vector_type(16))) __bf16 v16bf;
typedef __attribute__((ext_vector_type(8)))  float  v8f;
typedef __attribute__((ext_vector_type(4)))  int    v4i;

#define B_   2
#define T_   2048
#define H_   768
#define NH_  12
#define HD_  64
#define H3_  2304
#define BT_  (B_*T_)

union FragB16 { v16bf v; uint4 q[2]; };

// ---------------------------------------------------------------------------
// Async global->LDS copy (16B per lane), guarded so it degrades to a sync
// copy if the builtin is absent.  ASYNCcnt-tracked on gfx1250.
// Builtin signature (from clang diagnostic): (v4i AS1*, v4i AS3*, imm, imm).
// ---------------------------------------------------------------------------
#if defined(__HIP_DEVICE_COMPILE__) && __has_builtin(__builtin_amdgcn_global_load_async_to_lds_b128)
#define ASYNC_G2L 1
#else
#define ASYNC_G2L 0
#endif

#if ASYNC_G2L
using g_v4i = __attribute__((address_space(1))) v4i*;
using l_v4i = __attribute__((address_space(3))) v4i*;
#endif

__device__ __forceinline__ void g2l_b128(const bf16* g, bf16* l) {
#if ASYNC_G2L
    __builtin_amdgcn_global_load_async_to_lds_b128((g_v4i)g, (l_v4i)l, 0, 0);
#else
    *(uint4*)l = *(const uint4*)g;
#endif
}

__device__ __forceinline__ void wait_async0() {
#if ASYNC_G2L
 #if __has_builtin(__builtin_amdgcn_s_wait_asynccnt)
    __builtin_amdgcn_s_wait_asynccnt(0);
 #else
    asm volatile("s_wait_asynccnt 0" ::: "memory");
 #endif
#endif
}

// ---------------------------------------------------------------------------
// fp32 -> bf16 conversion (vectorized, grid-stride)
// ---------------------------------------------------------------------------
__global__ __launch_bounds__(256)
void cvt_f32_bf16(const float* __restrict__ src, bf16* __restrict__ dst, int n) {
    int i = (blockIdx.x * blockDim.x + threadIdx.x) * 4;
    int stride = gridDim.x * blockDim.x * 4;
    for (; i < n; i += stride) {
        float4 f = *(const float4*)(src + i);
        union { bf16 h[4]; uint2 u; } o;
        o.h[0] = (bf16)f.x; o.h[1] = (bf16)f.y;
        o.h[2] = (bf16)f.z; o.h[3] = (bf16)f.w;
        *(uint2*)(dst + i) = o.u;
    }
}

// ---------------------------------------------------------------------------
// Tiled WMMA GEMM: C[M,N] = A[M,K](bf16) * B[K,N](bf16) + bias[N]
// 128x128 block tile, BK=32, 8 waves (2x4), each wave 64x32 via 4x2 WMMA tiles
// Double-buffered LDS; A staged with async global->LDS, B transposed manually.
// ---------------------------------------------------------------------------
template<bool OUT_F32>
__global__ __launch_bounds__(256)
void gemm_bf16(const bf16* __restrict__ A, const bf16* __restrict__ Bm,
               const float* __restrict__ bias, void* __restrict__ Cout,
               int M, int N, int K) {
    __shared__ bf16 As[2][128 * 40];    // row-major, padded stride 40
    __shared__ bf16 Bts[2][128 * 40];   // transposed: Bts[n][k], padded stride 40

    const int tid  = threadIdx.x;
    const int wid  = tid >> 5;
    const int lane = tid & 31;
    const int m0 = blockIdx.x * 128;
    const int n0 = blockIdx.y * 128;
    const int wm = (wid >> 2) * 64;   // wave row offset (0/64)
    const int wn = (wid & 3) * 32;    // wave col offset (0..96)

    v8f acc[4][2] = {};

    const int arow = tid >> 1, aseg = (tid & 1) << 4;   // A staging coords
    const int bk   = tid & 31, bn   = (tid >> 5) << 4;  // B staging coords

    const int am  = lane & 15, ak  = (lane >> 4) << 3;  // A frag per-lane
    const int bnl = lane & 15, bkb = (lane >> 4) << 3;  // B frag per-lane

    auto stageA = [&](int buf, int kt) {
        const bf16* g = A + (size_t)(m0 + arow) * K + kt + aseg;
        g2l_b128(g,     &As[buf][arow * 40 + aseg]);
        g2l_b128(g + 8, &As[buf][arow * 40 + aseg + 8]);
    };
    auto stageB = [&](int buf, int kt) {
        const bf16* g = Bm + (size_t)(kt + bk) * N + n0 + bn;
        union { uint4 q[2]; bf16 h[16]; } t;
        t.q[0] = *(const uint4*)g;
        t.q[1] = *(const uint4*)(g + 8);
#pragma unroll
        for (int j = 0; j < 16; ++j)
            Bts[buf][(bn + j) * 40 + bk] = t.h[j];
    };

    // prologue: stage tile 0
    stageA(0, 0);
    stageB(0, 0);
    wait_async0();
    __syncthreads();

    const int nk = K >> 5;
    for (int ik = 0; ik < nk; ++ik) {
        const int cur = ik & 1;
        if (ik + 1 < nk) {              // issue next tile before computing
            stageA(cur ^ 1, (ik + 1) << 5);
            stageB(cur ^ 1, (ik + 1) << 5);
        }

        FragB16 a[4], b[2];
#pragma unroll
        for (int i = 0; i < 4; ++i) {
            const bf16* p = &As[cur][(wm + i * 16 + am) * 40 + ak];
            a[i].q[0] = *(const uint4*)p;         // K = h*8 .. h*8+7
            a[i].q[1] = *(const uint4*)(p + 16);  // K = 16+h*8 ..
        }
#pragma unroll
        for (int j = 0; j < 2; ++j) {
            const bf16* p = &Bts[cur][(wn + j * 16 + bnl) * 40 + bkb];
            b[j].q[0] = *(const uint4*)p;         // K = h*8 .. h*8+7
            b[j].q[1] = *(const uint4*)(p + 16);  // K = 16+h*8 ..
        }
#pragma unroll
        for (int i = 0; i < 4; ++i)
#pragma unroll
            for (int j = 0; j < 2; ++j)
                acc[i][j] = __builtin_amdgcn_wmma_f32_16x16x32_bf16(
                    false, a[i].v, false, b[j].v, (short)0, acc[i][j], false, false);

        wait_async0();
        __syncthreads();
    }

    // C/D layout: vgpr t, lanes 0-15 -> M=t, lanes 16-31 -> M=t+8; N = lane&15
    const int cn = lane & 15, moff = (lane >> 4) << 3;
#pragma unroll
    for (int i = 0; i < 4; ++i)
#pragma unroll
        for (int j = 0; j < 2; ++j) {
            int gn = n0 + wn + j * 16 + cn;
            float bv = bias[gn];
            size_t base = (size_t)(m0 + wm + i * 16 + moff) * N + gn;
#pragma unroll
            for (int t = 0; t < 8; ++t) {
                float v = acc[i][j][t] + bv;
                if (OUT_F32) ((float*)Cout)[base + (size_t)t * N] = v;
                else         ((bf16*)Cout)[base + (size_t)t * N] = (bf16)v;
            }
        }
}

// ---------------------------------------------------------------------------
// Flash attention: block = 128 queries x 1 (b,h); 8 waves, 16 queries each.
// Key blocks of 32, K/V staged in LDS (K async), online softmax, WMMA GEMMs.
// qkv layout: [B*T, 3H] bf16 (q at col h*64, k at 768+h*64, v at 1536+h*64)
// ---------------------------------------------------------------------------
__global__ __launch_bounds__(256)
void flash_attn(const bf16* __restrict__ qkv, const int* __restrict__ amask,
                bf16* __restrict__ outb) {
    const int q0 = blockIdx.x * 128;
    const int b  = blockIdx.y / NH_;
    const int h  = blockIdx.y % NH_;
    const int tid = threadIdx.x, wid = tid >> 5, lane = tid & 31;

    __shared__ bf16 Qs[128 * 72];      // Qs[qrow][d], padded
    __shared__ bf16 Ks[32 * 72];       // Ks[key][d]  (B-frag: contiguous d)
    __shared__ bf16 Vt[64 * 40];       // Vt[d][key]  (B-frag: contiguous key)
    __shared__ bf16 Ps[8 * 16 * 40];   // per-wave P scratch (C-layout -> A-frag)

    const bf16* qkv_b = qkv + (size_t)b * T_ * H3_;
    const int qcol = h * HD_, kcol = H_ + h * HD_, vcol = 2 * H_ + h * HD_;

    {   // stage Q block once (async)
        int row = tid >> 1, seg = (tid & 1) * 32;
        const bf16* g = qkv_b + (size_t)(q0 + row) * H3_ + qcol + seg;
#pragma unroll
        for (int u = 0; u < 4; ++u)
            g2l_b128(g + u * 8, &Qs[row * 72 + seg + u * 8]);
    }

    v8f O[4] = {};
    float mrow[8], lrow[8];
#pragma unroll
    for (int t = 0; t < 8; ++t) { mrow[t] = -3.0e38f; lrow[t] = 0.f; }

    const int wq    = wid * 16;       // wave's query offset within block
    const int lhalf = lane >> 4;
    const int lmod  = lane & 15;
    const float scale = 0.125f;       // 1/sqrt(64)

    wait_async0();                    // Q in LDS before first barrier

    const int nkb = (q0 + 128) / 32;
    for (int kb = 0; kb < nkb; ++kb) {
        const int ks = kb * 32;
        __syncthreads();
        {   // stage K tile 32x64 (async)
            int row = tid >> 3, seg = (tid & 7) * 8;
            const bf16* g = qkv_b + (size_t)(ks + row) * H3_ + kcol + seg;
            g2l_b128(g, &Ks[row * 72 + seg]);
        }
        {   // stage V tile transposed: Vt[d][key]
            int key = tid & 31, dseg = (tid >> 5) * 8;
            const bf16* g = qkv_b + (size_t)(ks + key) * H3_ + vcol + dseg;
            union { uint4 q; bf16 h[8]; } t;
            t.q = *(const uint4*)g;
#pragma unroll
            for (int j = 0; j < 8; ++j)
                Vt[(dseg + j) * 40 + key] = t.h[j];
        }
        wait_async0();
        __syncthreads();

        if (ks <= q0 + wq + 15) {   // skip fully-masked key blocks for this wave
            // S(16x32) = Q(16x64) @ K^T : two chained K=32 WMMAs per 16-key frag
            v8f s[2] = {};
#pragma unroll
            for (int kk = 0; kk < 64; kk += 32) {
                FragB16 aq;
                const bf16* ap = &Qs[(wq + lmod) * 72 + kk + lhalf * 8];
                aq.q[0] = *(const uint4*)ap;
                aq.q[1] = *(const uint4*)(ap + 16);
#pragma unroll
                for (int f = 0; f < 2; ++f) {
                    FragB16 bkf;
                    const bf16* bp = &Ks[(f * 16 + lmod) * 72 + kk + lhalf * 8];
                    bkf.q[0] = *(const uint4*)bp;
                    bkf.q[1] = *(const uint4*)(bp + 16);
                    s[f] = __builtin_amdgcn_wmma_f32_16x16x32_bf16(
                        false, aq.v, false, bkf.v, (short)0, s[f], false, false);
                }
            }
            // causal + key mask, online softmax (rows striped over 16 lanes)
            int key0 = ks + lmod, key1 = ks + 16 + lmod;
            int ok0 = amask[b * T_ + key0];
            int ok1 = amask[b * T_ + key1];
            bf16* pw = &Ps[wid * 640];
#pragma unroll
            for (int t = 0; t < 8; ++t) {
                int q = q0 + wq + lhalf * 8 + t;
                float s0 = (key0 <= q && ok0) ? s[0][t] * scale : -3.0e38f;
                float s1 = (key1 <= q && ok1) ? s[1][t] * scale : -3.0e38f;
                float mx = fmaxf(s0, s1);
#pragma unroll
                for (int d = 1; d < 16; d <<= 1)
                    mx = fmaxf(mx, __shfl_xor(mx, d));
                float mn = fmaxf(mrow[t], mx);
                float alpha = (mrow[t] <= -1.0e37f) ? 0.f : __expf(mrow[t] - mn);
                float p0 = (s0 <= -1.0e37f) ? 0.f : __expf(s0 - mn);
                float p1 = (s1 <= -1.0e37f) ? 0.f : __expf(s1 - mn);
                float psum = p0 + p1;
#pragma unroll
                for (int d = 1; d < 16; d <<= 1)
                    psum += __shfl_xor(psum, d);
                lrow[t] = lrow[t] * alpha + psum;
                mrow[t] = mn;
#pragma unroll
                for (int j = 0; j < 4; ++j) O[j][t] *= alpha;
                pw[(lhalf * 8 + t) * 40 + lmod]      = (bf16)p0;
                pw[(lhalf * 8 + t) * 40 + 16 + lmod] = (bf16)p1;
            }
            // O(16x64) += P(16x32) @ V(32x64): 4 WMMAs, V frags from Vt
            FragB16 pa;
            const bf16* pp = &Ps[wid * 640 + lmod * 40 + lhalf * 8];
            pa.q[0] = *(const uint4*)pp;
            pa.q[1] = *(const uint4*)(pp + 16);
#pragma unroll
            for (int j = 0; j < 4; ++j) {
                FragB16 vb;
                const bf16* vp = &Vt[(j * 16 + lmod) * 40 + lhalf * 8];
                vb.q[0] = *(const uint4*)vp;
                vb.q[1] = *(const uint4*)(vp + 16);
                O[j] = __builtin_amdgcn_wmma_f32_16x16x32_bf16(
                    false, pa.v, false, vb.v, (short)0, O[j], false, false);
            }
        }
    }

    // normalize and store bf16 attention output [B*T, H]
#pragma unroll
    for (int t = 0; t < 8; ++t) {
        float rl = lrow[t] > 0.f ? 1.f / lrow[t] : 0.f;
        int q = q0 + wq + lhalf * 8 + t;
#pragma unroll
        for (int j = 0; j < 4; ++j)
            outb[((size_t)b * T_ + q) * H_ + h * HD_ + j * 16 + lmod] =
                (bf16)(O[j][t] * rl);
    }
}

// ---------------------------------------------------------------------------
// Launch: convert -> QKV GEMM -> flash attention -> projection GEMM
// ---------------------------------------------------------------------------
extern "C" void kernel_launch(void* const* d_in, const int* in_sizes, int n_in,
                              void* d_out, int out_size, void* d_ws, size_t ws_size,
                              hipStream_t stream) {
    (void)in_sizes; (void)n_in; (void)out_size; (void)ws_size;
    const float* x     = (const float*)d_in[0];
    const int*   amask = (const int*)  d_in[1];
    const float* Wqkv  = (const float*)d_in[2];
    const float* bqkv  = (const float*)d_in[3];
    const float* Wproj = (const float*)d_in[4];
    const float* bproj = (const float*)d_in[5];
    float* out = (float*)d_out;

    char* ws = (char*)d_ws;
    size_t off = 0;
    auto carve = [&](size_t bytes) -> char* {
        char* p = ws + off;
        off = (off + bytes + 255) & ~(size_t)255;
        return p;
    };
    bf16* xb     = (bf16*)carve((size_t)BT_ * H_  * 2);
    bf16* wqkvb  = (bf16*)carve((size_t)H_  * H3_ * 2);
    bf16* wprojb = (bf16*)carve((size_t)H_  * H_  * 2);
    bf16* qkvb   = (bf16*)carve((size_t)BT_ * H3_ * 2);
    bf16* attnb  = (bf16*)carve((size_t)BT_ * H_  * 2);

    cvt_f32_bf16<<<1024, 256, 0, stream>>>(x,     xb,     BT_ * H_);
    cvt_f32_bf16<<<512,  256, 0, stream>>>(Wqkv,  wqkvb,  H_ * H3_);
    cvt_f32_bf16<<<256,  256, 0, stream>>>(Wproj, wprojb, H_ * H_);

    gemm_bf16<false><<<dim3(BT_ / 128, H3_ / 128), 256, 0, stream>>>(
        xb, wqkvb, bqkv, qkvb, BT_, H3_, H_);

    flash_attn<<<dim3(T_ / 128, B_ * NH_), 256, 0, stream>>>(qkvb, amask, attnb);

    gemm_bf16<true><<<dim3(BT_ / 128, H_ / 128), 256, 0, stream>>>(
        attnb, wprojb, bproj, out, BT_, H_, H_);
}
